// LinearAttentionHead_32272384262421
// MI455X (gfx1250) — matrix operand order repaired and checked by reading, not compiled, not run on hardware
//
#include <hip/hip_runtime.h>
#include <hip/hip_bf16.h>

#define BB 16
#define NN 2048
#define DD 128
#define BM 64          // query rows per workgroup (4 waves x 16)
#define BKV 32         // kv columns per inner tile
#define SCALE 0.08838834764831845f   // 1/sqrt(128)

typedef __attribute__((ext_vector_type(16))) _Float16 v16h;
typedef __attribute__((ext_vector_type(8)))  _Float16 v8h;
typedef __attribute__((ext_vector_type(8)))  float    v8f;

static __device__ __forceinline__ v16h cat16(v8h lo, v8h hi) {
    return __builtin_shufflevector(lo, hi, 0,1,2,3,4,5,6,7,8,9,10,11,12,13,14,15);
}

// ---------------- preprocessing ----------------

__global__ void bias_sum_kernel(const float* __restrict__ a, const float* __restrict__ b,
                                const float* __restrict__ c, const float* __restrict__ d,
                                float* __restrict__ o) {
    size_t i = (size_t)blockIdx.x * blockDim.x + threadIdx.x;
    o[i] = a[i] + b[i] + c[i] + d[i];
}

__global__ void qk_convert_kernel(const float* __restrict__ Q, const float* __restrict__ K,
                                  const int* __restrict__ emb_mask, const int* __restrict__ in_mask,
                                  _Float16* __restrict__ Qh, _Float16* __restrict__ Kh) {
    size_t i = (size_t)blockIdx.x * blockDim.x + threadIdx.x;   // over B*N*D
    size_t bn = i / DD;
    float q = emb_mask[bn] ? Q[i] : 0.0f;
    float k = in_mask[bn]  ? K[i] : 0.0f;
    Qh[i] = (_Float16)q;
    Kh[i] = (_Float16)k;
}

// V (B,N,D) fp32 -> Vt (B,D,N) f16, with input mask applied. 32x32 LDS tiles.
__global__ void v_transpose_kernel(const float* __restrict__ V, const int* __restrict__ in_mask,
                                   _Float16* __restrict__ Vt) {
    __shared__ float tile[32][33];
    const int b  = blockIdx.z;
    const int n0 = blockIdx.x * 32;
    const int d0 = blockIdx.y * 32;
    const int tx = threadIdx.x, ty = threadIdx.y;
    #pragma unroll
    for (int i = ty; i < 32; i += 8) {
        const int n = n0 + i;
        float v = V[((size_t)b * NN + n) * DD + d0 + tx];
        tile[i][tx] = in_mask[b * NN + n] ? v : 0.0f;
    }
    __syncthreads();
    #pragma unroll
    for (int i = ty; i < 32; i += 8) {
        Vt[((size_t)b * DD + d0 + i) * NN + n0 + tx] = (_Float16)tile[tx][i];
    }
}

// ---------------- flash attention ----------------

__global__ __launch_bounds__(128) void attn_kernel(
    const _Float16* __restrict__ Qh,   // (B,N,D) f16, emb-masked
    const _Float16* __restrict__ Kh,   // (B,N,D) f16, in-masked
    const _Float16* __restrict__ Vt,   // (B,D,N) f16, in-masked
    const float*    __restrict__ biasS,// (N,N) f32 summed bias
    float*          __restrict__ out)  // (B,N,D) f32
{
    __shared__ _Float16 Plds[4][16][32];   // per-wave P staging

    const int b     = blockIdx.y;
    const int qbase = blockIdx.x * BM;
    const int wave  = threadIdx.x >> 5;
    const int lane  = threadIdx.x & 31;
    const int hi    = lane >> 4;          // lane group 0/1
    const int ln    = lane & 15;

    const int qw = qbase + wave * 16;     // this wave's first query row

    // Q A-fragments: 4 slices of K=32 along D.
    // A k order used here (same as B): group hi holds K = hi*16 .. hi*16+15 contiguous.
    const _Float16* qrow = Qh + ((size_t)b * NN + qw + ln) * DD;
    v16h qa[4];
    #pragma unroll
    for (int ks = 0; ks < 4; ++ks) {
        const _Float16* p = qrow + ks * 32 + hi * 16;
        qa[ks] = cat16(*(const v8h*)p, *(const v8h*)(p + 8));
    }

    v8f acc[8];
    #pragma unroll
    for (int f = 0; f < 8; ++f) acc[f] = (v8f){};
    float mrow[8], lrow[8];
    #pragma unroll
    for (int r = 0; r < 8; ++r) { mrow[r] = -3.0e38f; lrow[r] = 0.0f; }

    const int ntiles = (qw + 15) / BKV + 1;   // causal: kv0 <= qw+15

    for (int t = 0; t < ntiles; ++t) {
        const int kv0 = t * BKV;

        // ---- S = Q K^T : two 16x16 fragments over kv columns ----
        v8f S0 = (v8f){}, S1 = (v8f){};
        // B k order used here (same as A): lane group hi holds K = hi*16 .. hi*16+15 contiguous.
        const _Float16* kr0 = Kh + ((size_t)b * NN + kv0 + ln) * DD;        // cols kv0..kv0+15
        const _Float16* kr1 = kr0 + 16 * DD;                                // cols kv0+16..kv0+31
        #pragma unroll
        for (int ks = 0; ks < 4; ++ks) {
            const _Float16* p0 = kr0 + ks * 32 + hi * 16;
            v16h b0 = cat16(*(const v8h*)p0, *(const v8h*)(p0 + 8));
            S0 = __builtin_amdgcn_wmma_f32_16x16x32_f16(false, qa[ks], false, b0,
                                                        (short)0, S0, false, false);
            const _Float16* p1 = kr1 + ks * 32 + hi * 16;
            v16h b1 = cat16(*(const v8h*)p1, *(const v8h*)(p1 + 8));
            S1 = __builtin_amdgcn_wmma_f32_16x16x32_f16(false, qa[ks], false, b1,
                                                        (short)0, S1, false, false);
        }

        // ---- scale + bias + causal mask + online softmax (C layout) ----
        // VGPR r, lane group hi -> row r + 8*hi ; col = ln (+16 for S1)
        #pragma unroll
        for (int r = 0; r < 8; ++r) {
            const int q = qw + r + 8 * hi;
            float s0 = S0[r] * SCALE + biasS[(size_t)q * NN + kv0 + ln];
            float s1 = S1[r] * SCALE + biasS[(size_t)q * NN + kv0 + 16 + ln];
            if (kv0 + ln      > q) s0 = -3.0e38f;
            if (kv0 + 16 + ln > q) s1 = -3.0e38f;

            float tmax = fmaxf(s0, s1);
            tmax = fmaxf(tmax, __shfl_xor(tmax, 1));
            tmax = fmaxf(tmax, __shfl_xor(tmax, 2));
            tmax = fmaxf(tmax, __shfl_xor(tmax, 4));
            tmax = fmaxf(tmax, __shfl_xor(tmax, 8));
            const float mnew = fmaxf(mrow[r], tmax);
            const float sc   = __expf(mrow[r] - mnew);
            mrow[r] = mnew;

            const float e0 = __expf(s0 - mnew);
            const float e1 = __expf(s1 - mnew);
            float rs = e0 + e1;
            rs += __shfl_xor(rs, 1);
            rs += __shfl_xor(rs, 2);
            rs += __shfl_xor(rs, 4);
            rs += __shfl_xor(rs, 8);
            lrow[r] = lrow[r] * sc + rs;

            #pragma unroll
            for (int f = 0; f < 8; ++f) acc[f][r] *= sc;

            Plds[wave][r + 8 * hi][ln]      = (_Float16)e0;
            Plds[wave][r + 8 * hi][ln + 16] = (_Float16)e1;
        }

        // ---- read P back as a 16x32 A-fragment (per-wave LDS, in-order) ----
        const _Float16* pl = &Plds[wave][ln][hi * 16];
        v16h pa = cat16(*(const v8h*)pl, *(const v8h*)(pl + 8));

        // ---- O += P V : 8 d-fragments, B from transposed V (contiguous) ----
        #pragma unroll
        for (int f = 0; f < 8; ++f) {
            const _Float16* vp = Vt + ((size_t)b * DD + f * 16 + ln) * NN + kv0 + hi * 16;
            v16h vb = cat16(*(const v8h*)vp, *(const v8h*)(vp + 8));
            acc[f] = __builtin_amdgcn_wmma_f32_16x16x32_f16(false, pa, false, vb,
                                                            (short)0, acc[f], false, false);
        }
    }

    // ---- normalize and store ----
    float invl[8];
    #pragma unroll
    for (int r = 0; r < 8; ++r) invl[r] = 1.0f / lrow[r];
    #pragma unroll
    for (int f = 0; f < 8; ++f) {
        #pragma unroll
        for (int r = 0; r < 8; ++r) {
            out[((size_t)b * NN + qw + r + 8 * hi) * DD + f * 16 + ln] = acc[f][r] * invl[r];
        }
    }
}

// ---------------- launcher ----------------

extern "C" void kernel_launch(void* const* d_in, const int* in_sizes, int n_in,
                              void* d_out, int out_size, void* d_ws, size_t ws_size,
                              hipStream_t stream) {
    const float* Q  = (const float*)d_in[0];
    const float* K  = (const float*)d_in[1];
    const float* V  = (const float*)d_in[2];
    const float* b1 = (const float*)d_in[3];
    const float* b2 = (const float*)d_in[4];
    const float* b3 = (const float*)d_in[5];
    const float* b4 = (const float*)d_in[6];
    const int* in_mask  = (const int*)d_in[7];
    const int* emb_mask = (const int*)d_in[8];
    // d_in[9] = causal_mask: not needed (computed analytically)

    float* out = (float*)d_out;
    char* ws = (char*)d_ws;
    float*    biasS = (float*)ws;                                  // 16 MB
    _Float16* Qh    = (_Float16*)(ws + (size_t)16 * 1024 * 1024);  //  8 MB
    _Float16* Kh    = (_Float16*)(ws + (size_t)24 * 1024 * 1024);  //  8 MB
    _Float16* Vt    = (_Float16*)(ws + (size_t)32 * 1024 * 1024);  //  8 MB

    bias_sum_kernel<<<(NN * NN) / 256, 256, 0, stream>>>(b1, b2, b3, b4, biasS);
    qk_convert_kernel<<<(BB * NN * DD) / 256, 256, 0, stream>>>(Q, K, emb_mask, in_mask, Qh, Kh);
    dim3 gt(NN / 32, DD / 32, BB);
    dim3 bt(32, 8);
    v_transpose_kernel<<<gt, bt, 0, stream>>>(V, in_mask, Vt);
    dim3 ga(NN / BM, BB);
    attn_kernel<<<ga, 128, 0, stream>>>(Qh, Kh, Vt, biasS, out);
}
